// MambaFusionLayer_52304111730833
// MI455X (gfx1250) — matrix operand order repaired and checked by reading, not compiled, not run on hardware
//
#include <hip/hip_runtime.h>

// ---------------------------------------------------------------------------
// MambaFusionLayer for MI455X (gfx1250):
//  - WMMA f16 GEMMs, register-blocked 16x64 per wave (A-fragment reuse x4)
//  - selective scan with ASYNC global->LDS staging of B_t/C_t (ASYNCcnt)
//  - out_proj / LN2 / head computed only for the 256 rows that reach d_out
// ---------------------------------------------------------------------------

typedef __attribute__((ext_vector_type(16))) _Float16 v16h;
typedef __attribute__((ext_vector_type(8)))  float    v8f;

#define D_MODEL 512
#define D_INNER 1024
#define D_STATE 256
#define DT_RANK 32
#define DBL_W   (DT_RANK + 2 * D_STATE)   // 544
#define LX      2048
#define NFUSED  64
#define LTOT    (LX + NFUSED)             // 2112
#define BATCH   4
#define MROWS   (BATCH * LTOT)            // 8448
#define MLAST   (BATCH * NFUSED)          // 256
#define EPSLN   1e-5f

#define MODE_XZ   0
#define MODE_DBL  1
#define MODE_DT   2
#define MODE_OUT  3
#define MODE_HEAD 4

typedef __attribute__((address_space(3))) float lds_float;

// ---------------------------------------------------------------- utilities
__global__ void cvt_f32_to_f16_kernel(const float* __restrict__ in,
                                      _Float16* __restrict__ out, int n) {
    int i = blockIdx.x * blockDim.x + threadIdx.x;
    if (i < n) out[i] = (_Float16)in[i];
}

__global__ void concat_kernel(const float* __restrict__ x,
                              const float* __restrict__ tok,
                              float* __restrict__ h) {
    size_t idx = (size_t)blockIdx.x * blockDim.x + threadIdx.x;
    if (idx >= (size_t)MROWS * D_MODEL) return;
    int c   = (int)(idx % D_MODEL);
    int row = (int)(idx / D_MODEL);
    int t = row % LTOT, b = row / LTOT;
    h[idx] = (t < LX) ? x[((size_t)b * LX + t) * D_MODEL + c]
                      : tok[(size_t)(t - LX) * D_MODEL + c];
}

// one wave per token: 512 columns -> 16 per lane, shfl_xor reduction (wave32)
__global__ void ln_kernel(const float* __restrict__ in,
                          const float* __restrict__ g,
                          const float* __restrict__ bta,
                          _Float16* __restrict__ out, int ntok) {
    int lane = threadIdx.x & 31;
    int tok  = blockIdx.x * (blockDim.x >> 5) + (threadIdx.x >> 5);
    if (tok >= ntok) return;
    const float* p = in + (size_t)tok * D_MODEL;
    float vals[16], s = 0.f, ss = 0.f;
#pragma unroll
    for (int k = 0; k < 16; ++k) {
        float v = p[lane + 32 * k];
        vals[k] = v; s += v; ss += v * v;
    }
#pragma unroll
    for (int off = 16; off; off >>= 1) {
        s  += __shfl_xor(s,  off, 32);
        ss += __shfl_xor(ss, off, 32);
    }
    float mean = s * (1.f / D_MODEL);
    float var  = ss * (1.f / D_MODEL) - mean * mean;
    float rstd = rsqrtf(var + EPSLN);
#pragma unroll
    for (int k = 0; k < 16; ++k) {
        int c = lane + 32 * k;
        out[(size_t)tok * D_MODEL + c] =
            (_Float16)((vals[k] - mean) * rstd * g[c] + bta[c]);
    }
}

// ------------------------------------------------------------- WMMA GEMM
// out = A[M,K] x Bw[N,K]^T. Each wave owns a 16x64 strip: one A fragment is
// reused across 4 N-tiles (4 accumulators, 4 WMMAs per 32-wide K step).
// Fragment layouts per CDNA5 ISA 7.12.2 (16-bit A/B, f32 C/D).
__global__ void wmma_gemm_kernel(const _Float16* __restrict__ A,
                                 const _Float16* __restrict__ Bw,
                                 int M, int N, int K, int mode,
                                 float* __restrict__ out0,
                                 float* __restrict__ out1,
                                 _Float16* __restrict__ out16,
                                 const float* __restrict__ bias,
                                 const float* __restrict__ resid) {
    int lane = threadIdx.x & 31;
    int wave = threadIdx.x >> 5;
    int ntn  = N >> 4;                 // N tiles
    int ngrp = (ntn + 3) >> 2;         // groups of 4 N-tiles
    int grp  = blockIdx.x * 4 + wave;
    if (grp >= (M >> 4) * ngrp) return;
    int m0  = (grp / ngrp) << 4;
    int ng0 = (grp % ngrp) << 2;       // first N-tile of the strip

    int halfsel = lane >> 4;
    const unsigned* rowA =
        (const unsigned*)(A + (size_t)(m0 + (lane & 15)) * K);
    const unsigned* rowB[4];
#pragma unroll
    for (int q = 0; q < 4; ++q) {
        int nt = (ng0 + q < ntn) ? (ng0 + q) : ng0;   // clamp edge group
        rowB[q] = (const unsigned*)(Bw + (size_t)((nt << 4) + (lane & 15)) * K);
    }

    union Frag { v16h h; unsigned u[8]; };
    v8f c[4] = {};
    for (int kb = 0; kb < K; kb += 32) {
        Frag fa;
        const unsigned* pa = rowA + (kb >> 1);
#pragma unroll
        for (int j = 0; j < 8; ++j)
            // A 16x32: lanes 0-15 hold K 0-7 & 16-23, lanes 16-31 hold +8
            fa.u[j] = pa[(j < 4 ? j : j + 4) + 4 * halfsel];
#pragma unroll
        for (int q = 0; q < 4; ++q) {
            Frag fb;
            const unsigned* pb = rowB[q] + (kb >> 1);
#pragma unroll
            for (int j = 0; j < 8; ++j)
                // B 32x16: lanes 0-15 hold K 0-7 & 16-23, lanes 16-31 hold +8
                fb.u[j] = pb[(j < 4 ? j : j + 4) + 4 * halfsel];
            c[q] = __builtin_amdgcn_wmma_f32_16x16x32_f16(
                       false, fa.h, false, fb.h, (short)0, c[q], false, false);
        }
    }

#pragma unroll
    for (int q = 0; q < 4; ++q) {
        if (ng0 + q >= ntn) continue;
        int col = ((ng0 + q) << 4) + (lane & 15);
#pragma unroll
        for (int r = 0; r < 8; ++r) {
            int row = m0 + 8 * halfsel + r;
            float v = c[q][r];
            if (mode == MODE_XZ) {
                if (col < D_INNER) {
                    out0[(size_t)row * D_INNER + col] = v;       // u pre-conv
                } else {
                    int t = row % LTOT, b = row / LTOT;
                    if (t >= LX)                                  // z, last 64
                        out1[((size_t)(b * NFUSED + (t - LX))) * D_INNER +
                             (col - D_INNER)] = v;
                }
            } else if (mode == MODE_DBL) {
                out0[(size_t)row * DBL_W + col] = v;              // dt|B|C
                if (col < DT_RANK)
                    out16[(size_t)row * DT_RANK + col] = (_Float16)v;
            } else if (mode == MODE_DT) {
                float xx = v + bias[col];
                float sp = (xx > 20.f) ? xx : log1pf(__expf(xx)); // softplus
                out0[(size_t)row * D_INNER + col] = sp;           // delta
            } else if (mode == MODE_OUT) {
                int hrow = (row / NFUSED) * LTOT + LX + (row % NFUSED);
                out0[(size_t)row * D_MODEL + col] =
                    v + resid[(size_t)hrow * D_MODEL + col];      // + h
            } else { // MODE_HEAD
                out0[(size_t)row * D_MODEL + col] =
                    v + bias[col] + resid[(size_t)row * D_MODEL + col];
            }
        }
    }
}

// ------------------------------------------------- depthwise causal conv+silu
__global__ void conv_silu_kernel(const float* __restrict__ uPre,
                                 const float* __restrict__ Wc,
                                 const float* __restrict__ bc,
                                 float* __restrict__ u32o,
                                 _Float16* __restrict__ u16o) {
    size_t idx = (size_t)blockIdx.x * blockDim.x + threadIdx.x;
    if (idx >= (size_t)MROWS * D_INNER) return;
    int d   = (int)(idx % D_INNER);
    int row = (int)(idx / D_INNER);
    int t = row % LTOT;
    float acc = bc[d];
#pragma unroll
    for (int k = 0; k < 4; ++k) {
        int tt = t - 3 + k;
        if (tt >= 0)
            acc += uPre[idx + (size_t)(k - 3) * D_INNER] * Wc[d * 4 + k];
    }
    float s = acc / (1.f + __expf(-acc));    // silu
    u32o[idx] = s;
    u16o[idx] = (_Float16)s;
}

// --------------------------------------------------------- selective scan
// Block = 8 waves (256 thr) = one batch x 8 channels. Lane owns 8 states
// (n = lane*8+j) in VGPRs. B_t/C_t staged into double-buffered LDS with
// CDNA5 async global->LDS copies (ASYNCcnt), one s_barrier per step.
// y reduced (shfl_xor) only for the last-64 window that reaches the output.
__device__ __forceinline__ void async_copy_f32_to_lds(float* lds_dst,
                                                      const float* gsrc) {
    unsigned ldsa = (unsigned)(unsigned long long)(lds_float*)lds_dst;
    unsigned long long ga = (unsigned long long)gsrc;
    asm volatile("global_load_async_to_lds_b32 %0, %1, off"
                 :: "v"(ldsa), "v"(ga) : "memory");
}

__global__ void scan_kernel(const float* __restrict__ u,
                            const float* __restrict__ delta,
                            const float* __restrict__ dbl,
                            const float* __restrict__ A_log,
                            float* __restrict__ y) {
    __shared__ float sB[2][D_STATE];
    __shared__ float sC[2][D_STATE];
    int b    = blockIdx.x >> 7;      // / 128
    int dblk = blockIdx.x & 127;
    int wave = threadIdx.x >> 5;
    int lane = threadIdx.x & 31;
    int d = dblk * 8 + wave;

    float Ar[8], hs[8];
#pragma unroll
    for (int j = 0; j < 8; ++j) {
        Ar[j] = -__expf(A_log[(size_t)d * D_STATE + lane * 8 + j]);
        hs[j] = 0.f;
    }
    const float* dblb = dbl + (size_t)b * LTOT * DBL_W;

    async_copy_f32_to_lds(&sB[0][threadIdx.x], dblb + DT_RANK + threadIdx.x);
    async_copy_f32_to_lds(&sC[0][threadIdx.x],
                          dblb + DT_RANK + D_STATE + threadIdx.x);
    asm volatile("s_wait_asynccnt 0" ::: "memory");
    __syncthreads();

    for (int t = 0; t < LTOT; ++t) {
        int cur = t & 1, nxt = cur ^ 1;
        if (t + 1 < LTOT) {
            const float* src = dblb + (size_t)(t + 1) * DBL_W;
            async_copy_f32_to_lds(&sB[nxt][threadIdx.x],
                                  src + DT_RANK + threadIdx.x);
            async_copy_f32_to_lds(&sC[nxt][threadIdx.x],
                                  src + DT_RANK + D_STATE + threadIdx.x);
            if (t + 2 < LTOT)   // -> global_prefetch_b8
                __builtin_prefetch(
                    dblb + (size_t)(t + 2) * DBL_W + DT_RANK + threadIdx.x,
                    0, 1);
        }
        float dt_ = delta[((size_t)(b * LTOT + t)) * D_INNER + d];
        float ut  = u[((size_t)(b * LTOT + t)) * D_INNER + d];
        float du  = dt_ * ut;
        float acc = 0.f;
#pragma unroll
        for (int j = 0; j < 8; ++j) {
            float dA = __expf(dt_ * Ar[j]);                  // v_exp_f32
            hs[j] = dA * hs[j] + du * sB[cur][lane * 8 + j];
            acc = fmaf(hs[j], sC[cur][lane * 8 + j], acc);
        }
        if (t >= LX) {       // output only needed for the fused-token window
#pragma unroll
            for (int off = 16; off; off >>= 1) acc += __shfl_xor(acc, off, 32);
            if (lane == 0)
                y[((size_t)(b * NFUSED + (t - LX))) * D_INNER + d] = acc;
        }
        asm volatile("s_wait_asynccnt 0" ::: "memory");
        __syncthreads();
    }
}

// ------------------------------------------------ Dp skip + silu(z) gating
__global__ void gate_kernel(const float* __restrict__ y,
                            const float* __restrict__ u32,
                            const float* __restrict__ z,
                            const float* __restrict__ Dp,
                            _Float16* __restrict__ yg) {
    int idx = blockIdx.x * blockDim.x + threadIdx.x;
    if (idx >= MLAST * D_INNER) return;
    int d   = idx % D_INNER;
    int row = idx / D_INNER;          // b*64 + i
    int b = row >> 6, i = row & 63;
    float uv = u32[((size_t)b * LTOT + LX + i) * D_INNER + d];
    float zv = z[idx];
    float yv = y[idx] + uv * Dp[d];
    float sz = zv / (1.f + __expf(-zv));
    yg[idx] = (_Float16)(yv * sz);
}

// ---------------------------------------------------------------------------
extern "C" void kernel_launch(void* const* d_in, const int* in_sizes, int n_in,
                              void* d_out, int out_size, void* d_ws, size_t ws_size,
                              hipStream_t stream) {
    const float* x      = (const float*)d_in[0];
    const float* tok    = (const float*)d_in[1];
    const float* g1     = (const float*)d_in[2];
    const float* b1     = (const float*)d_in[3];
    const float* g2     = (const float*)d_in[4];
    const float* b2     = (const float*)d_in[5];
    const float* W_head = (const float*)d_in[6];
    const float* b_head = (const float*)d_in[7];
    const float* W_in   = (const float*)d_in[8];
    const float* W_conv = (const float*)d_in[9];
    const float* b_conv = (const float*)d_in[10];
    const float* W_xp   = (const float*)d_in[11];
    const float* W_dt   = (const float*)d_in[12];
    const float* b_dt   = (const float*)d_in[13];
    const float* A_log  = (const float*)d_in[14];
    const float* Dp     = (const float*)d_in[15];
    const float* W_out  = (const float*)d_in[16];

    char* w = (char*)d_ws;
    auto carve = [&](size_t bytes) -> void* {
        void* p = (void*)w;
        w += (bytes + 255) & ~(size_t)255;
        return p;
    };
    float*    h        = (float*)   carve((size_t)MROWS * D_MODEL * 4);
    _Float16* a16      = (_Float16*)carve((size_t)MROWS * D_MODEL * 2);
    _Float16* W_in16   = (_Float16*)carve((size_t)2 * D_INNER * D_MODEL * 2);
    _Float16* W_xp16   = (_Float16*)carve((size_t)DBL_W * D_INNER * 2);
    _Float16* W_dt16   = (_Float16*)carve((size_t)D_INNER * DT_RANK * 2);
    _Float16* W_out16  = (_Float16*)carve((size_t)D_MODEL * D_INNER * 2);
    _Float16* W_head16 = (_Float16*)carve((size_t)D_MODEL * D_MODEL * 2);
    float*    uPre     = (float*)   carve((size_t)MROWS * D_INNER * 4);
    float*    z_last   = (float*)   carve((size_t)MLAST * D_INNER * 4);
    float*    u32b     = (float*)   carve((size_t)MROWS * D_INNER * 4);
    _Float16* u16b     = (_Float16*)carve((size_t)MROWS * D_INNER * 2);
    float*    dblb     = (float*)   carve((size_t)MROWS * DBL_W * 4);
    _Float16* dt16     = (_Float16*)carve((size_t)MROWS * DT_RANK * 2);
    float*    deltab   = (float*)   carve((size_t)MROWS * D_INNER * 4);
    float*    yb       = (float*)   carve((size_t)MLAST * D_INNER * 4);
    _Float16* yg16     = (_Float16*)carve((size_t)MLAST * D_INNER * 2);
    float*    h2       = (float*)   carve((size_t)MLAST * D_MODEL * 4);
    _Float16* ln216    = (_Float16*)carve((size_t)MLAST * D_MODEL * 2);

    auto gemm_grid = [](int M, int N) {
        int ngrp = ((N / 16) + 3) / 4;
        return ((M / 16) * ngrp + 3) / 4;
    };

    // 1. h = concat(x, fused_tokens)
    concat_kernel<<<(MROWS * D_MODEL + 255) / 256, 256, 0, stream>>>(x, tok, h);

    // 2. f16 weight copies
    cvt_f32_to_f16_kernel<<<(2 * D_INNER * D_MODEL + 255) / 256, 256, 0, stream>>>(W_in, W_in16, 2 * D_INNER * D_MODEL);
    cvt_f32_to_f16_kernel<<<(DBL_W * D_INNER + 255) / 256, 256, 0, stream>>>(W_xp, W_xp16, DBL_W * D_INNER);
    cvt_f32_to_f16_kernel<<<(D_INNER * DT_RANK + 255) / 256, 256, 0, stream>>>(W_dt, W_dt16, D_INNER * DT_RANK);
    cvt_f32_to_f16_kernel<<<(D_MODEL * D_INNER + 255) / 256, 256, 0, stream>>>(W_out, W_out16, D_MODEL * D_INNER);
    cvt_f32_to_f16_kernel<<<(D_MODEL * D_MODEL + 255) / 256, 256, 0, stream>>>(W_head, W_head16, D_MODEL * D_MODEL);

    // 3. a = LN1(h) -> f16
    ln_kernel<<<(MROWS + 3) / 4, 128, 0, stream>>>(h, g1, b1, a16, MROWS);

    // 4. xz = a @ W_in^T  (u pre-conv full; z only last-64 rows)
    wmma_gemm_kernel<<<gemm_grid(MROWS, 2 * D_INNER), 128, 0, stream>>>(
        a16, W_in16, MROWS, 2 * D_INNER, D_MODEL, MODE_XZ,
        uPre, z_last, nullptr, nullptr, nullptr);

    // 5. u = silu(causal dwconv(u))
    conv_silu_kernel<<<(MROWS * D_INNER + 255) / 256, 256, 0, stream>>>(
        uPre, W_conv, b_conv, u32b, u16b);

    // 6. dbl = u @ W_xp^T  (dt|B|C; dt also as f16)
    wmma_gemm_kernel<<<gemm_grid(MROWS, DBL_W), 128, 0, stream>>>(
        u16b, W_xp16, MROWS, DBL_W, D_INNER, MODE_DBL,
        dblb, nullptr, dt16, nullptr, nullptr);

    // 7. delta = softplus(dt @ W_dt^T + b_dt)
    wmma_gemm_kernel<<<gemm_grid(MROWS, D_INNER), 128, 0, stream>>>(
        dt16, W_dt16, MROWS, D_INNER, DT_RANK, MODE_DT,
        deltab, nullptr, nullptr, b_dt, nullptr);

    // 8. selective scan (y emitted only for fused-token window)
    scan_kernel<<<BATCH * (D_INNER / 8), 256, 0, stream>>>(
        u32b, deltab, dblb, A_log, yb);

    // 9. y = (y + u*Dp) * silu(z)   [last 64 tokens]
    gate_kernel<<<(MLAST * D_INNER + 255) / 256, 256, 0, stream>>>(
        yb, u32b, z_last, Dp, yg16);

    // 10. h2 = y @ W_out^T + h      [last 64 tokens]
    wmma_gemm_kernel<<<gemm_grid(MLAST, D_MODEL), 128, 0, stream>>>(
        yg16, W_out16, MLAST, D_MODEL, D_INNER, MODE_OUT,
        h2, nullptr, nullptr, nullptr, h);

    // 11. LN2(h2) -> f16
    ln_kernel<<<(MLAST + 3) / 4, 128, 0, stream>>>(h2, g2, b2, ln216, MLAST);

    // 12. out = LN2(h2) @ W_head^T + b_head + h2
    wmma_gemm_kernel<<<gemm_grid(MLAST, D_MODEL), 128, 0, stream>>>(
        ln216, W_head16, MLAST, D_MODEL, D_MODEL, MODE_HEAD,
        (float*)d_out, nullptr, nullptr, b_head, h2);
}
